// LmBMET_46694884442531
// MI455X (gfx1250) — matrix operand order repaired and checked by reading, not compiled, not run on hardware
//
#include <hip/hip_runtime.h>

// ---------------------------------------------------------------------------
// Transformer-XL layer for MI455X (gfx1250): all GEMMs via v_wmma_f32_16x16x32_bf16,
// B-tiles staged to LDS with GLOBAL_LOAD_ASYNC_TO_LDS_B128 (double-buffered).
// Q=512 M=512 K=1024 B=8 D=1024 H=16 DH=64 DI=4096
// ---------------------------------------------------------------------------

#define DEV __device__ __forceinline__

typedef __attribute__((ext_vector_type(16))) __bf16 v16bf;
typedef __attribute__((ext_vector_type(8)))  float  v8f;

union FragAB { v16bf v; unsigned int u[8]; };
union FragC  { v8f   v; float        f[8]; };

#if __has_builtin(__builtin_amdgcn_global_load_async_to_lds_b128) && \
    __has_builtin(__builtin_amdgcn_s_wait_asynccnt)
#define USE_ASYNC_LDS 1
#else
#define USE_ASYNC_LDS 0
#endif

DEV unsigned short f2bf(float x) {
    union { float f; unsigned int u; } c; c.f = x;
    unsigned int r = c.u + 0x7FFFu + ((c.u >> 16) & 1u);   // round-to-nearest-even
    return (unsigned short)(r >> 16);
}
DEV float bf2f(unsigned short h) {
    union { float f; unsigned int u; } c; c.u = ((unsigned int)h) << 16; return c.f;
}
DEV v8f wmma_bf16(v16bf a, v16bf b, v8f c) {
    // (neg_a, A, neg_b, B, c_mod, C, reuse_a, reuse_b)
    return __builtin_amdgcn_wmma_f32_16x16x32_bf16(false, a, false, b, (short)0, c, false, false);
}
// A-fragment dword 'vg' -> K offset (contiguous order: k = 16*half + 2*vg, matches B)
DEV int kmap(int vg, int khalf) { return 2*vg + 2*khalf; }
DEV FragC fzero() { FragC c;
#pragma unroll
    for (int j = 0; j < 8; ++j) c.f[j] = 0.f;
    return c; }

// 16B global -> LDS copy: async on gfx1250 (ASYNCcnt) with VGPR-bounce fallback
#if USE_ASYNC_LDS
typedef int v4i __attribute__((vector_size(16)));
typedef __attribute__((address_space(1))) v4i as1_v4i;
typedef __attribute__((address_space(3))) v4i as3_v4i;
#endif
DEV void copy_b128_to_lds(const unsigned short* g, unsigned short* l) {
#if USE_ASYNC_LDS
    __builtin_amdgcn_global_load_async_to_lds_b128(
        (as1_v4i*)g, (as3_v4i*)l, 0, 0);
#else
    *(uint4*)l = *(const uint4*)g;
#endif
}
DEV void async_wait0() {
#if USE_ASYNC_LDS
    __builtin_amdgcn_s_wait_asynccnt(0);
#endif
}

// ---------------------------------------------------------------------------
// fp32 -> bf16 bulk convert
// ---------------------------------------------------------------------------
__global__ void k_f32_to_bf16(const float* __restrict__ s, unsigned short* __restrict__ d, int n) {
    for (int i = blockIdx.x * blockDim.x + threadIdx.x; i < n; i += gridDim.x * blockDim.x)
        d[i] = f2bf(s[i]);
}

// ---------------------------------------------------------------------------
// Generic GEMM: C[M,N] = A[M,Kd] * Bm[N,Kd]^T   (A,Bm bf16 row-major)
// block = 256 thr = 8 waves; tile 128(M) x 64(N); wave -> 16 x 64 strip.
// B tile double-buffered in LDS via async copy; A fragments loaded directly
// (each half-lane's K-range is two contiguous b128s).
// ---------------------------------------------------------------------------
template<int BF16OUT, int BIAS, int RELU>
__global__ __launch_bounds__(256) void k_gemm(
    const unsigned short* __restrict__ A, int lda,
    const unsigned short* __restrict__ Bm, int ldb,
    void* __restrict__ Cv, int ldc, int Kd,
    const float* __restrict__ bias)
{
    __shared__ unsigned short ldsB[2][64][32];   // 8 KB ping-pong
    const int wave  = threadIdx.x >> 5;
    const int lane  = threadIdx.x & 31;
    const int khalf = (lane & 16) ? 8  : 0;
    const int bkoff = (lane & 16) ? 16 : 0;
    const int n0    = blockIdx.y * 64;
    const size_t arow = (size_t)(blockIdx.x * 128 + wave * 16 + (lane & 15)) * lda;

    const int sn  = threadIdx.x >> 2;            // staged column 0..63
    const int sch = threadIdx.x & 3;             // 16B chunk 0..3
    const unsigned short* bsrc = Bm + (size_t)(n0 + sn) * ldb + sch * 8;

    FragC acc[4];
#pragma unroll
    for (int t = 0; t < 4; ++t) acc[t] = fzero();

    copy_b128_to_lds(bsrc, &ldsB[0][sn][sch * 8]);            // prologue stage k0=0

    int cur = 0;
    for (int k0 = 0; k0 < Kd; k0 += 32, cur ^= 1) {
        async_wait0();
        __syncthreads();                                      // staged tile visible
        if (k0 + 32 < Kd) {
            copy_b128_to_lds(bsrc + k0 + 32, &ldsB[cur ^ 1][sn][sch * 8]);
            __builtin_prefetch(A + arow + k0 + 32, 0, 1);
        }

        FragAB fa;
        const unsigned short* ap = A + arow + k0;
#pragma unroll
        for (int vg = 0; vg < 8; ++vg) fa.u[vg] = *(const unsigned int*)(ap + kmap(vg, khalf));

#pragma unroll
        for (int t = 0; t < 4; ++t) {
            FragAB fb;
            const unsigned short* bp = &ldsB[cur][t * 16 + (lane & 15)][bkoff];
#pragma unroll
            for (int vg = 0; vg < 8; ++vg) fb.u[vg] = *(const unsigned int*)(bp + 2 * vg);
            acc[t].v = wmma_bf16(fa.v, fb.v, acc[t].v);
        }
    }

    const int ml   = (lane & 16) ? 8 : 0;
    const int mrow = blockIdx.x * 128 + wave * 16 + ml;
#pragma unroll
    for (int t = 0; t < 4; ++t) {
        const int col = n0 + t * 16 + (lane & 15);
        const float bv = BIAS ? bias[col] : 0.f;
#pragma unroll
        for (int j = 0; j < 8; ++j) {
            float v = acc[t].f[j] + bv;
            if (RELU) v = v > 0.f ? v : 0.f;
            const size_t off = (size_t)(mrow + j) * ldc + col;
            if (BF16OUT) ((unsigned short*)Cv)[off] = f2bf(v);
            else         ((float*)Cv)[off] = v;
        }
    }
}

// ---------------------------------------------------------------------------
// QKV GEMM: heads = cat(8192x1024) @ Wqkv(3072x1024)^T with scatter epilogue:
//   cols [0,1024)    -> q_bf  [(kpos-512)*8+b][c]          (only kpos>=512)
//   cols [1024,2048) -> kbf   [b][h][kpos][d]
//   cols [2048,3072) -> vtb   [b][h][d][kpos]   (transposed for P@V)
// ---------------------------------------------------------------------------
__global__ __launch_bounds__(256) void k_gemm_qkv(
    const unsigned short* __restrict__ A,
    const unsigned short* __restrict__ Bm,
    unsigned short* __restrict__ q_bf,
    unsigned short* __restrict__ kbf,
    unsigned short* __restrict__ vtb)
{
    __shared__ unsigned short ldsB[2][64][32];
    const int wave  = threadIdx.x >> 5;
    const int lane  = threadIdx.x & 31;
    const int khalf = (lane & 16) ? 8  : 0;
    const int bkoff = (lane & 16) ? 16 : 0;
    const int n0    = blockIdx.y * 64;
    const size_t arow = (size_t)(blockIdx.x * 128 + wave * 16 + (lane & 15)) * 1024;

    const int sn  = threadIdx.x >> 2;
    const int sch = threadIdx.x & 3;
    const unsigned short* bsrc = Bm + (size_t)(n0 + sn) * 1024 + sch * 8;

    FragC acc[4];
#pragma unroll
    for (int t = 0; t < 4; ++t) acc[t] = fzero();

    copy_b128_to_lds(bsrc, &ldsB[0][sn][sch * 8]);

    int cur = 0;
    for (int k0 = 0; k0 < 1024; k0 += 32, cur ^= 1) {
        async_wait0();
        __syncthreads();
        if (k0 + 32 < 1024) {
            copy_b128_to_lds(bsrc + k0 + 32, &ldsB[cur ^ 1][sn][sch * 8]);
            __builtin_prefetch(A + arow + k0 + 32, 0, 1);
        }

        FragAB fa;
        const unsigned short* ap = A + arow + k0;
#pragma unroll
        for (int vg = 0; vg < 8; ++vg) fa.u[vg] = *(const unsigned int*)(ap + kmap(vg, khalf));
#pragma unroll
        for (int t = 0; t < 4; ++t) {
            FragAB fb;
            const unsigned short* bp = &ldsB[cur][t * 16 + (lane & 15)][bkoff];
#pragma unroll
            for (int vg = 0; vg < 8; ++vg) fb.u[vg] = *(const unsigned int*)(bp + 2 * vg);
            acc[t].v = wmma_bf16(fa.v, fb.v, acc[t].v);
        }
    }

    const int ml   = (lane & 16) ? 8 : 0;
    const int mrow = blockIdx.x * 128 + wave * 16 + ml;
#pragma unroll
    for (int t = 0; t < 4; ++t) {
        const int col = n0 + t * 16 + (lane & 15);
        const int sec = col >> 10;           // 0=q 1=k 2=v
        const int hh  = (col & 1023) >> 6;
        const int dd  = col & 63;
#pragma unroll
        for (int j = 0; j < 8; ++j) {
            const int row  = mrow + j;
            const int kpos = row >> 3;       // cat row = kpos*B + b, B=8
            const int bb   = row & 7;
            const unsigned short val = f2bf(acc[t].f[j]);
            if (sec == 0) {
                if (kpos >= 512)
                    q_bf[(size_t)((kpos - 512) * 8 + bb) * 1024 + (col & 1023)] = val;
            } else if (sec == 1) {
                kbf[(((size_t)(bb * 16 + hh) * 1024 + kpos) << 6) + dd] = val;
            } else {
                vtb[(((size_t)(bb * 16 + hh) * 64 + dd) << 10) + kpos] = val;
            }
        }
    }
}

// ---------------------------------------------------------------------------
// Attention per (16-row q tile, b, h): BDraw->LDS, AC, rel-shift+mask, softmax,
// P@V, all matrix ops through WMMA. 128 threads = 4 waves.
// rel_shift: S[i][j] = (AC[i][j] + BDraw[i][j+511-i]) * 0.125, mask j > i+512.
// ---------------------------------------------------------------------------
__global__ __launch_bounds__(128) void k_attn(
    const unsigned short* __restrict__ q_bf,   // (512*8, 1024)
    const unsigned short* __restrict__ kbf,    // (8,16,1024,64)
    const unsigned short* __restrict__ vtb,    // (8,16,64,1024)
    const unsigned short* __restrict__ rk_bf,  // (1024, 1024) = (kpos, h*64+d)
    const float* __restrict__ rwb,             // (16,64)
    const float* __restrict__ rrb,             // (16,64)
    unsigned short* __restrict__ av_bf)        // (512*8, 1024)
{
    __shared__ unsigned short sbd[16][1024];   // 32 KB: BDraw, then scores, then P
    __shared__ float sred[16][8];

    const int qt   = blockIdx.x;               // 0..31
    const int bh   = blockIdx.y;               // 0..127
    const int b    = bh >> 4, h = bh & 15;
    const int wave = threadIdx.x >> 5;
    const int lane = threadIdx.x & 31;
    const int i0   = qt * 16;
    const int m    = lane & 15;
    const int khalf = (lane & 16) ? 8  : 0;
    const int bkoff = (lane & 16) ? 16 : 0;
    const int ml    = (lane & 16) ? 8  : 0;

    // --- biased q A-fragments (d inner, 64 = 2 K-steps), q+rwb and q+rrb ----
    const unsigned short* qrow = q_bf + ((size_t)(i0 + m) * 8 + b) * 1024 + h * 64;
    FragAB fqw[2], fqr[2];
#pragma unroll
    for (int ks = 0; ks < 2; ++ks) {
#pragma unroll
        for (int vg = 0; vg < 8; ++vg) {
            const int kk = ks * 32 + kmap(vg, khalf);
            const float x0 = bf2f(qrow[kk]), x1 = bf2f(qrow[kk + 1]);
            unsigned int w0 = f2bf(x0 + rwb[h * 64 + kk]);
            unsigned int w1 = f2bf(x1 + rwb[h * 64 + kk + 1]);
            fqw[ks].u[vg] = w0 | (w1 << 16);
            unsigned int r0 = f2bf(x0 + rrb[h * 64 + kk]);
            unsigned int r1 = f2bf(x1 + rrb[h * 64 + kk + 1]);
            fqr[ks].u[vg] = r0 | (r1 << 16);
        }
    }

    // --- Phase A: BDraw[16][1024] = (q+rrb) @ rk^T -> LDS (bf16) -----------
    for (int t = wave; t < 64; t += 4) {
        FragC acc = fzero();
#pragma unroll
        for (int ks = 0; ks < 2; ++ks) {
            FragAB fb;
            const unsigned short* bp = rk_bf + (size_t)(t * 16 + m) * 1024 + h * 64 + ks * 32 + bkoff;
#pragma unroll
            for (int vg = 0; vg < 8; ++vg) fb.u[vg] = *(const unsigned int*)(bp + 2 * vg);
            acc.v = wmma_bf16(fqr[ks].v, fb.v, acc.v);
        }
#pragma unroll
        for (int j = 0; j < 8; ++j) sbd[ml + j][t * 16 + m] = f2bf(acc.f[j]);
    }
    __syncthreads();

    // --- Phase B: AC + shifted BD + mask -> scores (regs), then -> LDS -----
    const int ntile = (qt + 33 < 64) ? (qt + 33) : 64;   // last tile touching j<=i+512
    FragC sc[16];
#pragma unroll
    for (int idx = 0; idx < 16; ++idx) {
        const int t = wave + idx * 4;                    // 0..63, always in range
        FragC acc = fzero();
#pragma unroll
        for (int ks = 0; ks < 2; ++ks) {
            FragAB fb;
            const unsigned short* bp = kbf + ((size_t)(b * 16 + h) * 1024 + t * 16 + m) * 64 + ks * 32 + bkoff;
#pragma unroll
            for (int vg = 0; vg < 8; ++vg) fb.u[vg] = *(const unsigned int*)(bp + 2 * vg);
            acc.v = wmma_bf16(fqw[ks].v, fb.v, acc.v);
        }
        const int jc = t * 16 + m;
#pragma unroll
        for (int j = 0; j < 8; ++j) {
            const int i = i0 + ml + j;
            float s = -1e30f;
            if (t < ntile && jc <= i + 512)
                s = (acc.f[j] + bf2f(sbd[ml + j][jc + 511 - i])) * 0.125f;
            sc[idx].f[j] = s;
        }
    }
    __syncthreads();               // all BD reads complete before overwriting
#pragma unroll
    for (int idx = 0; idx < 16; ++idx) {
        const int t = wave + idx * 4;
        if (t < ntile) {
            const int jc = t * 16 + m;
#pragma unroll
            for (int j = 0; j < 8; ++j) sbd[ml + j][jc] = f2bf(sc[idx].f[j]);
        }
    }
    __syncthreads();

    // --- softmax over rows (fp32 math, bf16 storage): 8 threads per row ----
    {
        const int row = threadIdx.x >> 3;
        const int sub = threadIdx.x & 7;
        const int lim = i0 + row + 512;
        const int jmax = lim < 1023 ? lim : 1023;
        float mx = -1e30f;
        for (int j = sub; j <= jmax; j += 8) mx = fmaxf(mx, bf2f(sbd[row][j]));
        sred[row][sub] = mx;
        __syncthreads();
        float rowmax = -1e30f;
#pragma unroll
        for (int s = 0; s < 8; ++s) rowmax = fmaxf(rowmax, sred[row][s]);
        float sum = 0.f;
        for (int j = sub; j <= jmax; j += 8) sum += __expf(bf2f(sbd[row][j]) - rowmax);
        __syncthreads();
        sred[row][sub] = sum;
        __syncthreads();
        float rowsum = 0.f;
#pragma unroll
        for (int s = 0; s < 8; ++s) rowsum += sred[row][s];
        const float inv = 1.f / rowsum;
        for (int j = sub; j < 1024; j += 8) {
            const float p = (j <= jmax) ? __expf(bf2f(sbd[row][j]) - rowmax) * inv : 0.f;
            sbd[row][j] = f2bf(p);     // masked/uncomputed region forced to 0
        }
    }
    __syncthreads();

    // --- Phase C: attn_vec tile = P(16x1024, LDS) @ V^T(vtb) ---------------
    {
        FragC acc = fzero();
        const unsigned short* vb = vtb + ((size_t)(b * 16 + h) * 64 + wave * 16 + m) * 1024 + bkoff;
        for (int k0 = 0; k0 < 1024; k0 += 32) {
            FragAB fa, fb;
            const unsigned short* ap = &sbd[m][k0];
#pragma unroll
            for (int vg = 0; vg < 8; ++vg) fa.u[vg] = *(const unsigned int*)(ap + kmap(vg, khalf));
#pragma unroll
            for (int vg = 0; vg < 8; ++vg) fb.u[vg] = *(const unsigned int*)(vb + k0 + 2 * vg);
            acc.v = wmma_bf16(fa.v, fb.v, acc.v);
        }
        const int col = h * 64 + wave * 16 + m;
#pragma unroll
        for (int j = 0; j < 8; ++j)
            av_bf[((size_t)(i0 + ml + j) * 8 + b) * 1024 + col] = f2bf(acc.f[j]);
    }
}

// ---------------------------------------------------------------------------
// LayerNorm(x + y) over D=1024, per row; writes fp32 and optional bf16 copy
// ---------------------------------------------------------------------------
__global__ __launch_bounds__(256) void k_ln(
    const float* __restrict__ x, const float* __restrict__ y,
    const float* __restrict__ g, const float* __restrict__ bta,
    float* __restrict__ of, unsigned short* __restrict__ obf)
{
    __shared__ float rs[256], rq[256];
    const int row = blockIdx.x, tid = threadIdx.x;
    const size_t base = (size_t)row * 1024;
    float s = 0.f, q = 0.f;
    for (int j = tid; j < 1024; j += 256) {
        const float v = x[base + j] + y[base + j];
        s += v; q += v * v;
    }
    rs[tid] = s; rq[tid] = q;
    __syncthreads();
    for (int off = 128; off > 0; off >>= 1) {
        if (tid < off) { rs[tid] += rs[tid + off]; rq[tid] += rq[tid + off]; }
        __syncthreads();
    }
    const float mu  = rs[0] * (1.f / 1024.f);
    const float var = rq[0] * (1.f / 1024.f) - mu * mu;
    const float inv = rsqrtf(var + 1e-5f);
    for (int j = tid; j < 1024; j += 256) {
        const float v = x[base + j] + y[base + j];
        const float o = (v - mu) * inv * g[j] + bta[j];
        of[base + j] = o;
        if (obf) obf[base + j] = f2bf(o);
    }
}

// ---------------------------------------------------------------------------
extern "C" void kernel_launch(void* const* d_in, const int* in_sizes, int n_in,
                              void* d_out, int out_size, void* d_ws, size_t ws_size,
                              hipStream_t stream)
{
    (void)in_sizes; (void)n_in; (void)out_size; (void)ws_size;
    const float* w    = (const float*)d_in[0];
    const float* r    = (const float*)d_in[1];
    const float* rwb  = (const float*)d_in[2];
    const float* rrb  = (const float*)d_in[3];
    const float* mems = (const float*)d_in[4];
    // d_in[5] = attn_mask (unused; mask computed analytically)
    const float* Wqkv = (const float*)d_in[6];
    const float* Wr   = (const float*)d_in[7];
    const float* Wo   = (const float*)d_in[8];
    const float* ln1g = (const float*)d_in[9];
    const float* ln1b = (const float*)d_in[10];
    const float* W1   = (const float*)d_in[11];
    const float* b1   = (const float*)d_in[12];
    const float* W2   = (const float*)d_in[13];
    const float* b2   = (const float*)d_in[14];
    const float* ln2g = (const float*)d_in[15];
    const float* ln2b = (const float*)d_in[16];

    char* ws = (char*)d_ws;
    size_t off = 0;
    auto alloc = [&](size_t bytes) -> char* {
        char* p = ws + off; off += (bytes + 255) & ~(size_t)255; return p;
    };
    unsigned short* cat_bf  = (unsigned short*)alloc(8192ull * 1024 * 2);
    unsigned short* wqkv_bf = (unsigned short*)alloc(3072ull * 1024 * 2);
    unsigned short* wr_bf   = (unsigned short*)alloc(1024ull * 1024 * 2);
    unsigned short* wo_bf   = (unsigned short*)alloc(1024ull * 1024 * 2);
    unsigned short* w1_bf   = (unsigned short*)alloc(4096ull * 1024 * 2);
    unsigned short* w2_bf   = (unsigned short*)alloc(4096ull * 1024 * 2);
    unsigned short* r_bf    = (unsigned short*)alloc(1024ull * 1024 * 2);
    unsigned short* q_bf    = (unsigned short*)alloc(4096ull * 1024 * 2);
    unsigned short* kbf     = (unsigned short*)alloc(8ull * 16 * 1024 * 64 * 2);
    unsigned short* vtb     = (unsigned short*)alloc(8ull * 16 * 64 * 1024 * 2);
    unsigned short* rk_bf   = (unsigned short*)alloc(1024ull * 1024 * 2);
    unsigned short* av_bf   = (unsigned short*)alloc(4096ull * 1024 * 2);
    float*          attn_o  = (float*)alloc(4096ull * 1024 * 4);
    float*          out1    = (float*)alloc(4096ull * 1024 * 4);
    unsigned short* out1_bf = (unsigned short*)alloc(4096ull * 1024 * 2);
    unsigned short* ff1_bf  = (unsigned short*)alloc(4096ull * 4096 * 2);
    float*          ff2     = (float*)alloc(4096ull * 1024 * 4);

    auto cvt = [&](const float* s, unsigned short* d, int n) {
        k_f32_to_bf16<<<dim3((n + 255) / 256), dim3(256), 0, stream>>>(s, d, n);
    };
    // cat = [mems ; w] flattened as (kpos*B+b, D)
    cvt(mems, cat_bf,                 512 * 8 * 1024);
    cvt(w,    cat_bf + 512 * 8 * 1024, 512 * 8 * 1024);
    cvt(Wqkv, wqkv_bf, 3072 * 1024);
    cvt(Wr,   wr_bf,   1024 * 1024);
    cvt(Wo,   wo_bf,   1024 * 1024);
    cvt(W1,   w1_bf,   4096 * 1024);
    cvt(W2,   w2_bf,   4096 * 1024);
    cvt(r,    r_bf,    1024 * 1024);

    // QKV projection with scatter epilogue
    k_gemm_qkv<<<dim3(64, 48), dim3(256), 0, stream>>>(cat_bf, wqkv_bf, q_bf, kbf, vtb);
    // r_k = r @ Wr^T  (bf16 out, (kpos, h*64+d))
    k_gemm<1, 0, 0><<<dim3(8, 16), dim3(256), 0, stream>>>(r_bf, 1024, wr_bf, 1024, rk_bf, 1024, 1024, nullptr);
    // attention
    k_attn<<<dim3(32, 128), dim3(128), 0, stream>>>(q_bf, kbf, vtb, rk_bf, rwb, rrb, av_bf);
    // attn_out = attn_vec @ Wo^T (fp32)
    k_gemm<0, 0, 0><<<dim3(32, 16), dim3(256), 0, stream>>>(av_bf, 1024, wo_bf, 1024, attn_o, 1024, 1024, nullptr);
    // out1 = LN(w + attn_out), also bf16 copy
    k_ln<<<dim3(4096), dim3(256), 0, stream>>>(w, attn_o, ln1g, ln1b, out1, out1_bf);
    // ff1 = relu(out1 @ W1^T + b1) (bf16)
    k_gemm<1, 1, 1><<<dim3(32, 64), dim3(256), 0, stream>>>(out1_bf, 1024, w1_bf, 1024, ff1_bf, 4096, 1024, b1);
    // ff2 = ff1 @ W2^T + b2 (fp32)
    k_gemm<0, 1, 0><<<dim3(32, 16), dim3(256), 0, stream>>>(ff1_bf, 4096, w2_bf, 4096, ff2, 1024, 4096, b2);
    // out = LN(out1 + ff2)
    k_ln<<<dim3(4096), dim3(256), 0, stream>>>(out1, ff2, ln2g, ln2b, (float*)d_out, nullptr);
}
